// ResidualCrossAttentionBlock_49890340110961
// MI455X (gfx1250) — matrix operand order repaired and checked by reading, not compiled, not run on hardware
//
#include <hip/hip_runtime.h>
#include <hip/hip_bf16.h>

typedef __attribute__((ext_vector_type(16))) _Float16 v16h;
typedef __attribute__((ext_vector_type(8)))  _Float16 v8h;
typedef __attribute__((ext_vector_type(8)))  float    v8f;

union AB { v16h v; v8h h[2]; };

#define Bx 16
#define Nx 1024
#define Hx 256
#define HEADSx 8
#define DHx 32
#define ROWS (Bx*Nx)        // 16384
#define F2H (2*Hx)          // 512

__device__ __forceinline__ v8f wmma16(const AB& a, const AB& b, v8f c) {
  return __builtin_amdgcn_wmma_f32_16x16x32_f16(
      /*neg_a=*/false, a.v, /*neg_b=*/false, b.v,
      /*c_mod=*/(short)0, c, /*reuse_a=*/false, /*reuse_b=*/false);
}

// ---------------------------------------------------------------- LayerNorm
__device__ __forceinline__ float block_sum256(float v, float* red, int t) {
  #pragma unroll
  for (int m = 16; m >= 1; m >>= 1) v += __shfl_xor(v, m, 32);
  __syncthreads();
  if ((t & 31) == 0) red[t >> 5] = v;
  __syncthreads();
  float s = red[t & 7];
  s += __shfl_xor(s, 4, 32);
  s += __shfl_xor(s, 2, 32);
  s += __shfl_xor(s, 1, 32);
  return s;
}

__global__ void ln_concat(const float* __restrict__ xs, const float* __restrict__ xo,
                          const float* __restrict__ gs, const float* __restrict__ bs,
                          const float* __restrict__ go, const float* __restrict__ bo,
                          _Float16* __restrict__ f_h) {
  __shared__ float red[8];
  int row = blockIdx.x;
  int t = threadIdx.x;                       // 0..255
  float a = xs[(size_t)row * Hx + t];
  float b = xo[(size_t)row * Hx + t];
  const float inv = 1.0f / (float)Hx;
  float ma = block_sum256(a, red, t) * inv;
  float mb = block_sum256(b, red, t) * inv;
  float da = a - ma, db = b - mb;
  float va = block_sum256(da * da, red, t) * inv;
  float vb = block_sum256(db * db, red, t) * inv;
  float na = da * rsqrtf(va + 1e-5f) * gs[t] + bs[t];
  float nb = db * rsqrtf(vb + 1e-5f) * go[t] + bo[t];
  f_h[(size_t)row * F2H + t]      = (_Float16)na;
  f_h[(size_t)row * F2H + Hx + t] = (_Float16)nb;
}

// ------------------------------------------------- weight f32[K][N] -> f16[N][K]
__global__ void cvt_transpose(const float* __restrict__ src, _Float16* __restrict__ dst,
                              int K, int Nn) {
  int idx = blockIdx.x * blockDim.x + threadIdx.x;
  if (idx < K * Nn) {
    int k = idx / Nn, n = idx % Nn;
    dst[(size_t)n * K + k] = (_Float16)src[idx];
  }
}

// ---------------------------------------------------------------- QKV GEMM
// f_h [16384,512] f16 x WT [256,512] f16 -> Q/K row-major f16, V transposed f16
__global__ void qkv_gemm(const _Float16* __restrict__ f_h,
                         const _Float16* __restrict__ WqT,
                         const _Float16* __restrict__ WkT,
                         const _Float16* __restrict__ WvT,
                         _Float16* __restrict__ Qh, _Float16* __restrict__ Kh,
                         _Float16* __restrict__ Vt) {
  int wave = threadIdx.x >> 5, lane = threadIdx.x & 31;
  int tile = blockIdx.x * 8 + wave;          // 0..16383
  int tm = tile >> 4, tn = tile & 15;
  int which = blockIdx.z;
  const _Float16* WT = (which == 0) ? WqT : (which == 1) ? WkT : WvT;
  int hh = lane >> 4, l15 = lane & 15;
  int kbA = hh * 8, kbB = hh * 16;
  const _Float16* arow = f_h + (size_t)(tm * 16 + l15) * F2H;
  const _Float16* brow = WT + (size_t)(tn * 16 + l15) * F2H;
  v8f c = {};
  #pragma unroll
  for (int k0 = 0; k0 < F2H; k0 += 32) {
    AB a, b;
    a.h[0] = *(const v8h*)(arow + k0 + kbB);
    a.h[1] = *(const v8h*)(arow + k0 + kbB + 8);
    b.h[0] = *(const v8h*)(brow + k0 + kbB);
    b.h[1] = *(const v8h*)(brow + k0 + kbB + 8);
    c = wmma16(a, b, c);
  }
  int rbase = tm * 16 + 8 * hh;
  int gcol = tn * 16 + l15;
  if (which == 2) {
    v8h o;
    #pragma unroll
    for (int r = 0; r < 8; r++) o[r] = (_Float16)c[r];
    *(v8h*)(Vt + (size_t)gcol * ROWS + rbase) = o;   // Vt[feature][token]
  } else {
    _Float16* O = (which == 0) ? Qh : Kh;
    #pragma unroll
    for (int r = 0; r < 8; r++) O[(size_t)(rbase + r) * Hx + gcol] = (_Float16)c[r];
  }
}

// ---------------------------------------------------------------- attention
// grid (16 qgroups, 8 heads, 16 batches), 128 threads = 4 waves, wave = one 16-query tile
__global__ void attn(const _Float16* __restrict__ Qh, const _Float16* __restrict__ Kh,
                     const _Float16* __restrict__ Vt, _Float16* __restrict__ Msg) {
  __shared__ __align__(16) _Float16 probs[4][16][32];
  int wave = threadIdx.x >> 5, lane = threadIdx.x & 31;
  int h = blockIdx.y, bz = blockIdx.z;
  int qt = blockIdx.x * 4 + wave;            // 0..63
  int hh = lane >> 4, l15 = lane & 15;
  int kbA = hh * 8, kbB = hh * 16;
  const float scale = 0.17677669529663687f;  // 1/sqrt(32)

  const _Float16* qrow = Qh + (size_t)(bz * Nx + qt * 16 + l15) * Hx + h * DHx;
  AB aq;
  aq.h[0] = *(const v8h*)(qrow + kbB);
  aq.h[1] = *(const v8h*)(qrow + kbB + 8);

  float m_run[8], l_run[8];
  v8f acc0 = {}, acc1 = {};
  #pragma unroll
  for (int r = 0; r < 8; r++) { m_run[r] = -1e30f; l_run[r] = 0.f; }

  for (int t = 0; t < Nx / 32; ++t) {
    int k0 = bz * Nx + t * 32;
    // ---- scores: S = Q (16x32) x K^T (32x16), two key tiles
    AB bk0, bk1;
    const _Float16* kp0 = Kh + (size_t)(k0 + l15) * Hx + h * DHx + kbB;
    const _Float16* kp1 = Kh + (size_t)(k0 + 16 + l15) * Hx + h * DHx + kbB;
    bk0.h[0] = *(const v8h*)(kp0); bk0.h[1] = *(const v8h*)(kp0 + 8);
    bk1.h[0] = *(const v8h*)(kp1); bk1.h[1] = *(const v8h*)(kp1 + 8);
    v8f s0 = {}, s1 = {};
    s0 = wmma16(aq, bk0, s0);
    s1 = wmma16(aq, bk1, s1);
    // ---- online softmax (rows live within 16-lane half-waves)
    #pragma unroll
    for (int r = 0; r < 8; r++) {
      float sa = s0[r] * scale, sb = s1[r] * scale;
      float mx = fmaxf(sa, sb);
      mx = fmaxf(mx, __shfl_xor(mx, 1, 32));
      mx = fmaxf(mx, __shfl_xor(mx, 2, 32));
      mx = fmaxf(mx, __shfl_xor(mx, 4, 32));
      mx = fmaxf(mx, __shfl_xor(mx, 8, 32));
      float mn = fmaxf(m_run[r], mx);
      float sc = __expf(m_run[r] - mn);
      float p0 = __expf(sa - mn), p1 = __expf(sb - mn);
      float ts = p0 + p1;
      ts += __shfl_xor(ts, 1, 32);
      ts += __shfl_xor(ts, 2, 32);
      ts += __shfl_xor(ts, 4, 32);
      ts += __shfl_xor(ts, 8, 32);
      l_run[r] = l_run[r] * sc + ts;
      m_run[r] = mn;
      acc0[r] *= sc;
      acc1[r] *= sc;
      int m = r + 8 * hh;
      probs[wave][m][l15]      = (_Float16)p0;
      probs[wave][m][16 + l15] = (_Float16)p1;
    }
    __syncthreads();  // uniform trip count across all 4 waves; orders LDS D->A transpose
    // ---- reload probabilities in A-operand layout (16 q x 32 keys)
    AB ap;
    ap.h[0] = *(const v8h*)(&probs[wave][l15][kbB]);
    ap.h[1] = *(const v8h*)(&probs[wave][l15][kbB + 8]);
    // ---- O += P (16x32) x V (32x32) as two 16-col WMMAs (Vt contiguous along keys)
    AB bv0, bv1;
    const _Float16* vp0 = Vt + (size_t)(h * DHx + l15) * ROWS + k0 + kbB;
    const _Float16* vp1 = Vt + (size_t)(h * DHx + 16 + l15) * ROWS + k0 + kbB;
    bv0.h[0] = *(const v8h*)(vp0); bv0.h[1] = *(const v8h*)(vp0 + 8);
    bv1.h[0] = *(const v8h*)(vp1); bv1.h[1] = *(const v8h*)(vp1 + 8);
    acc0 = wmma16(ap, bv0, acc0);
    acc1 = wmma16(ap, bv1, acc1);
    __syncthreads();
  }
  #pragma unroll
  for (int r = 0; r < 8; r++) {
    float rl = 1.0f / l_run[r];
    size_t tok = (size_t)bz * Nx + qt * 16 + r + 8 * hh;
    Msg[tok * Hx + h * DHx + l15]      = (_Float16)(acc0[r] * rl);
    Msg[tok * Hx + h * DHx + 16 + l15] = (_Float16)(acc1[r] * rl);
  }
}

// --------------------------------------------- output projection + residual
__global__ void outproj(const _Float16* __restrict__ Msg, const _Float16* __restrict__ WoT,
                        const float* __restrict__ x_out, float* __restrict__ x_new) {
  int wave = threadIdx.x >> 5, lane = threadIdx.x & 31;
  int tile = blockIdx.x * 8 + wave;
  int tm = tile >> 4, tn = tile & 15;
  int hh = lane >> 4, l15 = lane & 15;
  int kbA = hh * 8, kbB = hh * 16;
  const _Float16* arow = Msg + (size_t)(tm * 16 + l15) * Hx;
  const _Float16* brow = WoT + (size_t)(tn * 16 + l15) * Hx;
  v8f c = {};
  #pragma unroll
  for (int k0 = 0; k0 < Hx; k0 += 32) {
    AB a, b;
    a.h[0] = *(const v8h*)(arow + k0 + kbB);
    a.h[1] = *(const v8h*)(arow + k0 + kbB + 8);
    b.h[0] = *(const v8h*)(brow + k0 + kbB);
    b.h[1] = *(const v8h*)(brow + k0 + kbB + 8);
    c = wmma16(a, b, c);
  }
  int rbase = tm * 16 + 8 * hh;
  int gcol = tn * 16 + l15;
  #pragma unroll
  for (int r = 0; r < 8; r++) {
    size_t idx = (size_t)(rbase + r) * Hx + gcol;
    x_new[idx] = x_out[idx] + c[r];
  }
}

// --------------------------------------------- position head: p = (x_new - x_out) @ Wp
__global__ void pos_head(const float* __restrict__ x_new, const float* __restrict__ x_out,
                         const float* __restrict__ Wp, float* __restrict__ p_out) {
  int wave = threadIdx.x >> 5, lane = threadIdx.x & 31;
  int tok = blockIdx.x * 8 + wave;
  const float* xr = x_new + (size_t)tok * Hx;
  const float* xo = x_out + (size_t)tok * Hx;
  float a0 = 0.f, a1 = 0.f, a2 = 0.f;
  #pragma unroll
  for (int i = 0; i < 8; i++) {
    int f = lane * 8 + i;
    float mx = xr[f] - xo[f];
    a0 += mx * Wp[f * 3 + 0];
    a1 += mx * Wp[f * 3 + 1];
    a2 += mx * Wp[f * 3 + 2];
  }
  #pragma unroll
  for (int m = 16; m >= 1; m >>= 1) {
    a0 += __shfl_xor(a0, m, 32);
    a1 += __shfl_xor(a1, m, 32);
    a2 += __shfl_xor(a2, m, 32);
  }
  if (lane == 0) {
    p_out[(size_t)tok * 3 + 0] = a0;
    p_out[(size_t)tok * 3 + 1] = a1;
    p_out[(size_t)tok * 3 + 2] = a2;
  }
}

extern "C" void kernel_launch(void* const* d_in, const int* in_sizes, int n_in,
                              void* d_out, int out_size, void* d_ws, size_t ws_size,
                              hipStream_t stream) {
  const float* x_source = (const float*)d_in[0];
  // d_in[1] p_source unused by the reference math
  const float* x_out = (const float*)d_in[2];
  // d_in[3] p_out input unused
  const float* g_s = (const float*)d_in[4];
  const float* b_s = (const float*)d_in[5];
  const float* g_o = (const float*)d_in[6];
  const float* b_o = (const float*)d_in[7];
  const float* Wq = (const float*)d_in[8];
  const float* Wk = (const float*)d_in[9];
  const float* Wv = (const float*)d_in[10];
  const float* Wo = (const float*)d_in[11];
  const float* Wp = (const float*)d_in[12];

  float* x_new = (float*)d_out;                       // [16384, 256]
  float* p_new = x_new + (size_t)ROWS * Hx;           // [16384, 3]

  _Float16* f_h = (_Float16*)d_ws;                    // [16384, 512]
  _Float16* Qh  = f_h + (size_t)ROWS * F2H;           // [16384, 256]
  _Float16* Kh  = Qh + (size_t)ROWS * Hx;             // [16384, 256]
  _Float16* Vt  = Kh + (size_t)ROWS * Hx;             // [256, 16384]
  _Float16* Msg = Vt + (size_t)ROWS * Hx;             // [16384, 256]
  _Float16* WqT = Msg + (size_t)ROWS * Hx;            // [256, 512]
  _Float16* WkT = WqT + (size_t)F2H * Hx;
  _Float16* WvT = WkT + (size_t)F2H * Hx;
  _Float16* WoT = WvT + (size_t)F2H * Hx;             // [256, 256]

  ln_concat<<<ROWS, 256, 0, stream>>>(x_source, x_out, g_s, b_s, g_o, b_o, f_h);

  int nqkv = F2H * Hx;                                // 131072
  cvt_transpose<<<(nqkv + 255) / 256, 256, 0, stream>>>(Wq, WqT, F2H, Hx);
  cvt_transpose<<<(nqkv + 255) / 256, 256, 0, stream>>>(Wk, WkT, F2H, Hx);
  cvt_transpose<<<(nqkv + 255) / 256, 256, 0, stream>>>(Wv, WvT, F2H, Hx);
  int no = Hx * Hx;
  cvt_transpose<<<(no + 255) / 256, 256, 0, stream>>>(Wo, WoT, Hx, Hx);

  // 16384 output tiles per matrix, 8 waves/block
  qkv_gemm<<<dim3(2048, 1, 3), 256, 0, stream>>>(f_h, WqT, WkT, WvT, Qh, Kh, Vt);

  attn<<<dim3(16, HEADSx, Bx), 128, 0, stream>>>(Qh, Kh, Vt, Msg);

  outproj<<<2048, 256, 0, stream>>>(Msg, WoT, x_out, x_new);

  pos_head<<<2048, 256, 0, stream>>>(x_new, x_out, Wp, p_new);
}
